// SparseConvNet_64312840290846
// MI455X (gfx1250) — matrix operand order repaired and checked by reading, not compiled, not run on hardware
//
#include <hip/hip_runtime.h>

typedef __attribute__((ext_vector_type(16))) _Float16 v16h;
typedef __attribute__((ext_vector_type(8)))  _Float16 v8h;
typedef __attribute__((ext_vector_type(8)))  float    v8f;
typedef __attribute__((ext_vector_type(4)))  int      v4i;

#define BN_EPS 1e-3f

// CDNA5 async global->LDS staging (ASYNCcnt path), feature-gated so the file
// still compiles (register staging fallback) if the builtin is absent.
#if __has_builtin(__builtin_amdgcn_global_load_async_to_lds_b128)
#define HAVE_ASYNC_LDS 1
#else
#define HAVE_ASYNC_LDS 0
#endif

typedef __attribute__((address_space(1))) v4i gv4i;   // global (device) int4
typedef __attribute__((address_space(3))) v4i lv4i;   // LDS int4

__device__ __forceinline__ void wait_async_then_barrier() {
#if HAVE_ASYNC_LDS
#if __has_builtin(__builtin_amdgcn_s_wait_asynccnt)
  __builtin_amdgcn_s_wait_asynccnt(0);
#else
  asm volatile("s_wait_asynccnt 0x0" ::: "memory");
#endif
#endif
  __syncthreads();
}

// ---------------------------------------------------------------------------
// Pack input: x[P][C] * mask -> act[P][Cp] f16, zero channel padding.
// ---------------------------------------------------------------------------
__global__ void k_pack_input(const float* __restrict__ x, const float* __restrict__ mask,
                             _Float16* __restrict__ act, int P, int C, int Cp) {
  long tot = (long)P * Cp;
  for (long i = (long)blockIdx.x * blockDim.x + threadIdx.x; i < tot;
       i += (long)gridDim.x * blockDim.x) {
    int  c = (int)(i % Cp);
    long p = i / Cp;
    float v = (c < C) ? x[p * (long)C + c] * mask[p] : 0.f;
    act[i] = (_Float16)v;
  }
}

// ---------------------------------------------------------------------------
// Pack weights f32 [27][Cin][Cout] -> f16 WMMA-B order [Kchunk][Coutp][32],
// K = t*Cinp + ci, zero padding for ci>=Cin or co>=Cout.
// ---------------------------------------------------------------------------
__global__ void k_pack_w(const float* __restrict__ w, _Float16* __restrict__ wpk,
                         int Cin, int Cout, int Cinp, int Coutp) {
  long tot = 27L * Cinp * Coutp;
  for (long i = (long)blockIdx.x * blockDim.x + threadIdx.x; i < tot;
       i += (long)gridDim.x * blockDim.x) {
    int  co = (int)(i % Coutp);
    long K  = i / Coutp;               // 0 .. 27*Cinp-1
    int  ci = (int)(K % Cinp);
    int  t  = (int)(K / Cinp);
    long chunk = K >> 5;
    int  kin   = (int)(K & 31);
    float v = (ci < Cin && co < Cout) ? w[((long)t * Cin + ci) * Cout + co] : 0.f;
    wpk[(chunk * Coutp + co) * 32 + kin] = (_Float16)v;
  }
}

// ---------------------------------------------------------------------------
// Downsample mask: 3x3x3 max-pool, stride 2, pad 1.
// ---------------------------------------------------------------------------
__global__ void k_downmask(const float* __restrict__ mi, float* __restrict__ mo,
                           int Gi, int Go) {
  int i = blockIdx.x * blockDim.x + threadIdx.x;
  int tot = Go * Go * Go;
  if (i >= tot) return;
  int x = i % Go, y = (i / Go) % Go, z = i / (Go * Go);
  float m = 0.f;
  for (int tz = 0; tz < 3; tz++) {
    int zi = 2 * z - 1 + tz; if (zi < 0 || zi >= Gi) continue;
    for (int ty = 0; ty < 3; ty++) {
      int yi = 2 * y - 1 + ty; if (yi < 0 || yi >= Gi) continue;
      for (int tx = 0; tx < 3; tx++) {
        int xi = 2 * x - 1 + tx; if (xi < 0 || xi >= Gi) continue;
        m = fmaxf(m, mi[(zi * Gi + yi) * Gi + xi]);
      }
    }
  }
  mo[i] = (m > 0.f) ? 1.f : 0.f;
}

// ---------------------------------------------------------------------------
// Count active sites (single block, deterministic tree reduction).
// ---------------------------------------------------------------------------
__global__ void k_count(const float* __restrict__ m, float* __restrict__ out, int P) {
  __shared__ float s[256];
  float a = 0.f;
  for (int i = threadIdx.x; i < P; i += blockDim.x) a += m[i];
  s[threadIdx.x] = a;
  __syncthreads();
  for (int o = 128; o > 0; o >>= 1) {
    if ((int)threadIdx.x < o) s[threadIdx.x] += s[threadIdx.x + o];
    __syncthreads();
  }
  if (threadIdx.x == 0) *out = s[0];
}

// ---------------------------------------------------------------------------
// Masked per-channel sum / sumsq.  One block per channel; Cyp = row stride.
// ---------------------------------------------------------------------------
__global__ void k_stats(const float* __restrict__ y, const float* __restrict__ mask,
                        float* __restrict__ stats, int P, int Cyp) {
  __shared__ float s1[256], s2[256];
  int c = blockIdx.x;
  float a = 0.f, b = 0.f;
  for (int p = threadIdx.x; p < P; p += blockDim.x) {
    float m = mask[p];
    float v = y[(long)p * Cyp + c] * m;   // mask is exactly 0/1 -> v*v == y*y*m
    a += v;
    b += v * v;
  }
  s1[threadIdx.x] = a; s2[threadIdx.x] = b;
  __syncthreads();
  for (int o = 128; o > 0; o >>= 1) {
    if ((int)threadIdx.x < o) {
      s1[threadIdx.x] += s1[threadIdx.x + o];
      s2[threadIdx.x] += s2[threadIdx.x + o];
    }
    __syncthreads();
  }
  if (threadIdx.x == 0) { stats[2 * c] = s1[0]; stats[2 * c + 1] = s2[0]; }
}

// ---------------------------------------------------------------------------
// BN + ReLU + mask -> next-layer f16 activations.
// Cyp = conv-buffer row stride (read), Cap = activation row stride (write).
// ---------------------------------------------------------------------------
__global__ void k_apply(const float* __restrict__ y, const float* __restrict__ mask,
                        const float* __restrict__ stats, const float* __restrict__ nptr,
                        const float* __restrict__ gamma, const float* __restrict__ beta,
                        _Float16* __restrict__ act, int P, int Cout, int Cyp, int Cap) {
  float n = *nptr;
  long tot = (long)P * Cap;
  for (long i = (long)blockIdx.x * blockDim.x + threadIdx.x; i < tot;
       i += (long)gridDim.x * blockDim.x) {
    int  c = (int)(i % Cap);
    long p = i / Cap;
    float r = 0.f;
    if (c < Cout) {
      float mean = stats[2 * c] / n;
      float var  = stats[2 * c + 1] / n - mean * mean;
      float yn = (y[p * Cyp + c] - mean) * rsqrtf(var + BN_EPS) * gamma[c] + beta[c];
      r = fmaxf(yn, 0.f) * mask[p];
    }
    act[i] = (_Float16)r;
  }
}

// ---------------------------------------------------------------------------
// Final BN + ReLU + mask, write f32 transposed to NCDHW: out[c*P + p].
// ---------------------------------------------------------------------------
__global__ void k_apply_final(const float* __restrict__ y, const float* __restrict__ mask,
                              const float* __restrict__ stats, const float* __restrict__ nptr,
                              const float* __restrict__ gamma, const float* __restrict__ beta,
                              float* __restrict__ out, int P, int Cout) {
  float n = *nptr;
  long tot = (long)P * Cout;
  for (long i = (long)blockIdx.x * blockDim.x + threadIdx.x; i < tot;
       i += (long)gridDim.x * blockDim.x) {
    int c = (int)(i % Cout);
    int p = (int)(i / Cout);
    float mean = stats[2 * c] / n;
    float var  = stats[2 * c + 1] / n - mean * mean;
    float yn = (y[i] - mean) * rsqrtf(var + BN_EPS) * gamma[c] + beta[c];
    out[(long)c * P + p] = fmaxf(yn, 0.f) * mask[p];
  }
}

// ---------------------------------------------------------------------------
// Implicit-GEMM 3x3x3 conv via V_WMMA_F32_16X16X32_F16 with async LDS staging.
//
// Block (8 waves) computes a 256-position x 64-outchannel tile.
// Wave (mi 0..3, ni 0..1) computes 64 pos x 32 cout = 8 WMMAs / K-step.
// Per K-step (one 3x3x3 tap x one 32-channel chunk):
//   - stage A slab [256 rows][32 K] (16KB) + B slab [64 cols][32 K] (4KB)
//     into LDS ping-pong buffers via GLOBAL_LOAD_ASYNC_TO_LDS_B128
//     (halo rows zero-filled with ds_store); ASYNCcnt waited before barrier.
//   - one __syncthreads per step; WMMA operands read from LDS.
// All template dims are compile-time -> no integer division in codegen.
// ---------------------------------------------------------------------------
template <int GI, int GO, int STRIDE, int CINP, int COUTP>
__global__ void __launch_bounds__(256)
k_conv_wmma(const _Float16* __restrict__ act, const _Float16* __restrict__ wpk,
            float* __restrict__ out) {
  constexpr int NCC   = CINP / 32;   // 32-wide K chunks per tap
  constexpr int NSTEP = 27 * NCC;
  constexpr int P     = GO * GO * GO;
  constexpr int MBLK  = P / 256;     // position-tiles per column of blocks

  __shared__ _Float16 lA[2][256 * 32];   // 2 x 16 KB
  __shared__ _Float16 lB[2][64 * 32];    // 2 x  4 KB

  const int tid  = threadIdx.x;
  const int blkM = blockIdx.x % MBLK;
  const int blkN = blockIdx.x / MBLK;
  const int NB   = blkN * 64;

  // per-thread coords of the output row this thread stages
  const int R  = blkM * 256 + tid;
  const int rx = R % GO, ry = (R / GO) % GO, rz = R / (GO * GO);

  auto stage = [&](int step, int buf) {
    const int t = step / NCC, cc = step % NCC;
    const int tx = t % 3, ty = (t / 3) % 3, tz = t / 9;
    const int xi = rx * STRIDE - 1 + tx;
    const int yi = ry * STRIDE - 1 + ty;
    const int zi = rz * STRIDE - 1 + tz;
    const bool v = (unsigned)xi < (unsigned)GI && (unsigned)yi < (unsigned)GI &&
                   (unsigned)zi < (unsigned)GI;
    const _Float16* src = act + (long)((zi * GI + yi) * GI + xi) * CINP + cc * 32;
    const int col = tid >> 2, seg = tid & 3;
    const long kchunk = (long)t * NCC + cc;
    const _Float16* bsrc = wpk + (kchunk * COUTP + NB + col) * 32 + seg * 8;

#if HAVE_ASYNC_LDS
    if (v) {
      gv4i* s1 = (gv4i*)src;
      lv4i* dA = (lv4i*)&lA[buf][tid * 32];
      __builtin_amdgcn_global_load_async_to_lds_b128(s1,     dA,     0, 0);
      __builtin_amdgcn_global_load_async_to_lds_b128(s1 + 1, dA + 1, 0, 0);
      __builtin_amdgcn_global_load_async_to_lds_b128(s1 + 2, dA + 2, 0, 0);
      __builtin_amdgcn_global_load_async_to_lds_b128(s1 + 3, dA + 3, 0, 0);
    } else {
      v8h z = {};
      v8h* d = (v8h*)&lA[buf][tid * 32];
      d[0] = z; d[1] = z; d[2] = z; d[3] = z;
    }
    __builtin_amdgcn_global_load_async_to_lds_b128(
        (gv4i*)bsrc, (lv4i*)&lB[buf][col * 32 + seg * 8], 0, 0);
#else
    v8h z = {};
    v8h d0 = z, d1 = z, d2 = z, d3 = z;
    if (v) {
      d0 = ((const v8h*)src)[0];
      d1 = ((const v8h*)src)[1];
      d2 = ((const v8h*)src)[2];
      d3 = ((const v8h*)src)[3];
    }
    v8h* dstA = (v8h*)&lA[buf][tid * 32];
    dstA[0] = d0; dstA[1] = d1; dstA[2] = d2; dstA[3] = d3;
    *(v8h*)&lB[buf][col * 32 + seg * 8] = *(const v8h*)bsrc;
#endif
  };

  const int wave = tid >> 5, lane = tid & 31;
  const int mi = wave >> 1, ni = wave & 1;
  const int half = lane >> 4, q = lane & 15;

  v8f c[4][2] = {};

  stage(0, 0);
  wait_async_then_barrier();

  for (int s = 0; s < NSTEP; s++) {
    const int buf = s & 1;
    if (s + 1 < NSTEP) stage(s + 1, buf ^ 1);

    // B fragments (lane covers column q of each 16-wide tile; K = 16h..16h+15)
    union { v16h v; v8h h[2]; } B0, B1;
    {
      const _Float16* bb = &lB[buf][(ni * 32 + q) * 32 + 16 * half];
      B0.h[0] = *(const v8h*)bb;
      B0.h[1] = *(const v8h*)(bb + 8);
      const _Float16* bb1 = bb + 16 * 32;
      B1.h[0] = *(const v8h*)bb1;
      B1.h[1] = *(const v8h*)(bb1 + 8);
    }
#pragma unroll
    for (int f = 0; f < 4; f++) {
      // A fragment (K = 16h..16h+15, matching B)
      union { v16h v; v8h h[2]; } A;
      const _Float16* ab = &lA[buf][(mi * 64 + f * 16 + q) * 32 + 16 * half];
      A.h[0] = *(const v8h*)ab;
      A.h[1] = *(const v8h*)(ab + 8);
      c[f][0] = __builtin_amdgcn_wmma_f32_16x16x32_f16(false, A.v, false, B0.v,
                                                       (short)0, c[f][0], false, false);
      c[f][1] = __builtin_amdgcn_wmma_f32_16x16x32_f16(false, A.v, false, B1.v,
                                                       (short)0, c[f][1], false, false);
    }
    wait_async_then_barrier();
  }

  // C/D layout: lane = column q (half selects M block of 8), VGPR r = row r+8*half.
  const long rowbase = (long)blkM * 256 + mi * 64 + 8 * half;
  const int  colb    = NB + ni * 32 + q;
#pragma unroll
  for (int f = 0; f < 4; f++) {
#pragma unroll
    for (int r = 0; r < 8; r++) {
      out[(rowbase + f * 16 + r) * COUTP + colb]      = c[f][0][r];
      out[(rowbase + f * 16 + r) * COUTP + colb + 16] = c[f][1][r];
    }
  }
}

// ---------------------------------------------------------------------------
extern "C" void kernel_launch(void* const* d_in, const int* in_sizes, int n_in,
                              void* d_out, int out_size, void* d_ws, size_t ws_size,
                              hipStream_t stream) {
  const float* x     = (const float*)d_in[0];
  const float* mask0 = (const float*)d_in[1];
  const float* w0  = (const float*)d_in[2],  *g0  = (const float*)d_in[3],  *b0  = (const float*)d_in[4];
  const float* wd0 = (const float*)d_in[5],  *gd0 = (const float*)d_in[6],  *bd0 = (const float*)d_in[7];
  const float* w1  = (const float*)d_in[8],  *g1  = (const float*)d_in[9],  *b1  = (const float*)d_in[10];
  const float* wd1 = (const float*)d_in[11], *gd1 = (const float*)d_in[12], *bd1 = (const float*)d_in[13];
  const float* w2  = (const float*)d_in[14], *g2  = (const float*)d_in[15], *b2  = (const float*)d_in[16];

  char* ws = (char*)d_ws;
  size_t off = 0;
  auto alloc = [&](size_t bytes) { size_t o = off; off += (bytes + 255) & ~(size_t)255; return o; };

  // reused buffers (total ~140 MB)
  _Float16* wpk   = (_Float16*)(ws + alloc(56623104));   // largest packed weights (w2)
  _Float16* actA  = (_Float16*)(ws + alloc(18874368));   // ping
  _Float16* actB  = (_Float16*)(ws + alloc(18874368));   // pong
  float*    conv  = (float*)   (ws + alloc(41943040));   // largest f32 conv out (32768x320)
  float*    mask1 = (float*)   (ws + alloc(4096 * 4));
  float*    mask2 = (float*)   (ws + alloc(512 * 4));
  float*    stats = (float*)   (ws + alloc(1024 * 2 * 4));
  float*    cnts  = (float*)   (ws + alloc(256));

  // masks & active-site counts
  k_downmask<<<16, 256, 0, stream>>>(mask0, mask1, 32, 16);
  k_downmask<<<2, 256, 0, stream>>>(mask1, mask2, 16, 8);
  k_count<<<1, 256, 0, stream>>>(mask0, cnts + 0, 32768);
  k_count<<<1, 256, 0, stream>>>(mask1, cnts + 1, 4096);
  k_count<<<1, 256, 0, stream>>>(mask2, cnts + 2, 512);

  // ---- conv0: SubM 264->264 on 32^3 (Cinp 288, conv-out stride 320) ----
  k_pack_input<<<2048, 256, 0, stream>>>(x, mask0, actA, 32768, 264, 288);
  k_pack_w<<<1024, 256, 0, stream>>>(w0, wpk, 264, 264, 288, 320);
  k_conv_wmma<32, 32, 1, 288, 320><<<128 * 5, 256, 0, stream>>>(actA, wpk, conv);
  k_stats<<<264, 256, 0, stream>>>(conv, mask0, stats, 32768, 320);
  k_apply<<<2048, 256, 0, stream>>>(conv, mask0, stats, cnts + 0, g0, b0, actB,
                                    32768, 264, 320, 288);

  // ---- down0: 264->512, stride 2, 32^3 -> 16^3 ----
  k_pack_w<<<2048, 256, 0, stream>>>(wd0, wpk, 264, 512, 288, 512);
  k_conv_wmma<32, 16, 2, 288, 512><<<16 * 8, 256, 0, stream>>>(actB, wpk, conv);
  k_stats<<<512, 256, 0, stream>>>(conv, mask1, stats, 4096, 512);
  k_apply<<<1024, 256, 0, stream>>>(conv, mask1, stats, cnts + 1, gd0, bd0, actA,
                                    4096, 512, 512, 512);

  // ---- conv1: SubM 512->512 on 16^3 ----
  k_pack_w<<<2048, 256, 0, stream>>>(w1, wpk, 512, 512, 512, 512);
  k_conv_wmma<16, 16, 1, 512, 512><<<16 * 8, 256, 0, stream>>>(actA, wpk, conv);
  k_stats<<<512, 256, 0, stream>>>(conv, mask1, stats, 4096, 512);
  k_apply<<<1024, 256, 0, stream>>>(conv, mask1, stats, cnts + 1, g1, b1, actB,
                                    4096, 512, 512, 512);

  // ---- down1: 512->1024, stride 2, 16^3 -> 8^3 ----
  k_pack_w<<<4096, 256, 0, stream>>>(wd1, wpk, 512, 1024, 512, 1024);
  k_conv_wmma<16, 8, 2, 512, 1024><<<2 * 16, 256, 0, stream>>>(actB, wpk, conv);
  k_stats<<<1024, 256, 0, stream>>>(conv, mask2, stats, 512, 1024);
  k_apply<<<512, 256, 0, stream>>>(conv, mask2, stats, cnts + 2, gd1, bd1, actA,
                                   512, 1024, 1024, 1024);

  // ---- conv2: SubM 1024->1024 on 8^3, then final BN -> d_out (NCDHW f32) ----
  k_pack_w<<<4096, 256, 0, stream>>>(w2, wpk, 1024, 1024, 1024, 1024);
  k_conv_wmma<8, 8, 1, 1024, 1024><<<2 * 16, 256, 0, stream>>>(actA, wpk, conv);
  k_stats<<<1024, 256, 0, stream>>>(conv, mask2, stats, 512, 1024);
  k_apply_final<<<2048, 256, 0, stream>>>(conv, mask2, stats, cnts + 2, g2, b2,
                                          (float*)d_out, 512, 1024);
}
